// Decoder_4724464025785
// MI455X (gfx1250) — matrix operand order repaired and checked by reading, not compiled, not run on hardware
//
#include <hip/hip_runtime.h>
#include <hip/hip_bf16.h>

// ---------------------------------------------------------------------------
// MI455X (gfx1250) LSTM decoder.
// B=32, T=32, LAT=256, OUT=129, CH=12, HID=512, IN=397, 4H=2048.
//
// Critical path: 32 outer iters x 32 LSTM steps = 1024 dependent
// [16,512]x[512,2048] GEMMs per half-batch -> persistent workgroup,
// v_wmma_f32_16x16x32_bf16, f32 accumulate, cell state in VGPRs.
// ---------------------------------------------------------------------------

typedef __attribute__((ext_vector_type(16))) __bf16  v16bf;
typedef __attribute__((ext_vector_type(8)))  float   v8f;
typedef __attribute__((ext_vector_type(4)))  unsigned int v4u;

union ABfrag { v16bf v; v4u q[2]; };

__device__ __forceinline__ unsigned short f2bf(float f) {
  unsigned int u = __builtin_bit_cast(unsigned int, f);
  return (unsigned short)((u + 0x7FFFu + ((u >> 16) & 1u)) >> 16);  // RNE
}

__device__ __forceinline__ float sigf(float x) {
  return 1.0f / (1.0f + __expf(-x));
}

// A-fragment K index (16-bit A 16x32, ISA 7.12.2): lanes 0-15 e->{0..7,16..23},
// lanes 16-31 e->{8..15,24..31}.  B-fragment uses the same K order as A:
// lanes 0-15 e->{0..7,16..23}, lanes 16-31 e->{8..15,24..31}.
__device__ __forceinline__ int kmapB(int lane, int e) {
  return ((lane >> 4) << 3) + e + ((e >> 3) << 3);
}

// ---------------------------------------------------------------------------
// Prep 1: xw_static[b][t][g] = b_ih[g]+b_hh[g] + latent[b]*W_ih[g,:256]
//                              + chords[b,t]*W_ih[g,385:397]
// ---------------------------------------------------------------------------
__global__ void k_prep_xw(const float* __restrict__ chords,
                          const float* __restrict__ latent,
                          const float* __restrict__ W_ih,
                          const float* __restrict__ b_ih,
                          const float* __restrict__ b_hh,
                          float* __restrict__ xw) {
  int idx = blockIdx.x * 256 + threadIdx.x;      // ((b*32+t)*2048 + g)
  int g  = idx & 2047;
  int bt = idx >> 11;
  int b  = bt >> 5;
  const float* wr  = W_ih + (size_t)g * 397;
  const float* lat = latent + b * 256;
  const float* ch  = chords + bt * 12;
  float s = b_ih[g] + b_hh[g];
  for (int k = 0; k < 256; k++) s += lat[k] * wr[k];
  for (int c = 0; c < 12;  c++) s += ch[c] * wr[385 + c];
  xw[idx] = s;
}

// ---------------------------------------------------------------------------
// Prep 2: pack W_hh (B[k,g] = W_hh[g,k]) into per-(nt,kt,lane) 32B fragments.
// nt in [0,128), kt in [0,16), lane in [0,32): 16 bf16 per lane.
// ---------------------------------------------------------------------------
__global__ void k_pack_bhh(const float* __restrict__ W_hh,
                           unsigned short* __restrict__ pB) {
  int idx = blockIdx.x * 256 + threadIdx.x;
  if (idx >= 128 * 16 * 32) return;
  int lane = idx & 31, kt = (idx >> 5) & 15, nt = idx >> 9;
  int n = nt * 16 + (lane & 15);
  unsigned short* o = pB + (size_t)idx * 16;
  for (int e = 0; e < 16; e++)
    o[e] = f2bf(W_hh[(size_t)n * 512 + kt * 32 + kmapB(lane, e)]);
}

// Prep 3: pack W_lin (B[k,n] = W_lin[n,k]); N padded 129 -> 144 with zeros.
__global__ void k_pack_blin(const float* __restrict__ W_lin,
                            unsigned short* __restrict__ pB) {
  int idx = blockIdx.x * 256 + threadIdx.x;
  if (idx >= 9 * 16 * 32) return;
  int lane = idx & 31, kt = (idx >> 5) & 15, nt = idx >> 9;
  int n = nt * 16 + (lane & 15);
  unsigned short* o = pB + (size_t)idx * 16;
  for (int e = 0; e < 16; e++)
    o[e] = (n < 129) ? f2bf(W_lin[(size_t)n * 512 + kt * 32 + kmapB(lane, e)])
                     : (unsigned short)0;
}

// Prep 4: WprevT[p][g] = W_ih[g][256+p]  (one-hot prev => column gather)
__global__ void k_wprev(const float* __restrict__ W_ih,
                        float* __restrict__ wprevT) {
  int idx = blockIdx.x * 256 + threadIdx.x;   // p*2048 + g
  if (idx >= 129 * 2048) return;
  int g = idx & 2047, p = idx >> 11;
  wprevT[idx] = W_ih[(size_t)g * 397 + 256 + p];
}

// ---------------------------------------------------------------------------
// Main persistent kernel: 2 blocks x 512 threads; block handles 16 batch rows.
// Wave w owns hidden-column tiles jt = 2w, 2w+1 across all 4 gates
// (8 accumulators); c stays in VGPRs; h/tanh(h) round-trip via bf16 LDS.
// ---------------------------------------------------------------------------
__global__ __launch_bounds__(512)
void k_lstm(const float* __restrict__ xw,
            const unsigned short* __restrict__ pBhh,
            const unsigned short* __restrict__ pBlin,
            const float* __restrict__ wprevT,
            const float* __restrict__ b_lin,
            float* __restrict__ out) {
  __shared__ unsigned short lds_h[16 * 512];    // bf16 h          (16 KB)
  __shared__ unsigned short lds_th[16 * 512];   // bf16 tanh(h)    (16 KB)
  __shared__ float lds_logit[16 * 144];         // logits           (9 KB)
  __shared__ int prevBuf[2][16 * 32];           // prev one-hot idx (4 KB)

  const int tid  = threadIdx.x;
  const int lane = tid & 31;
  const int w    = tid >> 5;        // wave id 0..15
  const int col  = lane & 15;       // N (or M for A-frag) within tile
  const int half = lane >> 4;
  const int b0   = blockIdx.x * 16; // batch offset
  const int jt0  = w * 2;

  for (int x = tid; x < 16 * 32; x += 512) prevBuf[0][x] = -1;
  __syncthreads();

  v8f cst[2];
  v8f hval[2];

#pragma unroll 1
  for (int i = 0; i < 32; i++) {                // outer sampling loop
    int* cur = prevBuf[i & 1];
    int* nxt = prevBuf[(i + 1) & 1];
    if (tid < 16) nxt[tid * 32] = -1;           // prev[:,0,:]=0 always
    for (int x = tid; x < 16 * 512; x += 512) { lds_h[x] = 0; lds_th[x] = 0; }
#pragma unroll
    for (int j = 0; j < 2; j++)
#pragma unroll
      for (int r = 0; r < 8; r++) cst[j][r] = 0.0f;
    __syncthreads();

#pragma unroll 1
    for (int t = 0; t < 32; t++) {              // LSTM time steps
      // ---- phase A: z = xw_static + Wprev[prev] + h @ W_hh^T -------------
      v8f acc[8];
      int prow[8];
#pragma unroll
      for (int r = 0; r < 8; r++) prow[r] = cur[(r + half * 8) * 32 + t];
#pragma unroll
      for (int a = 0; a < 8; a++) {             // a = G*2 + jt2
        const int G = a >> 1, jt2 = a & 1;
        const int g = G * 512 + (jt0 + jt2) * 16 + col;
#pragma unroll
        for (int r = 0; r < 8; r++) {
          const int b = b0 + r + half * 8;
          float v = xw[((size_t)(b * 32 + t)) * 2048 + g];
          const int p = prow[r];
          if (p >= 0) v += wprevT[(size_t)p * 2048 + g];
          acc[a][r] = v;
        }
      }
#pragma unroll 1
      for (int kt = 0; kt < 16; kt++) {
        ABfrag af;                              // A: bf16 h, lane = batch row
        const unsigned short* hp = lds_h + col * 512 + kt * 32;
        af.q[0] = *(const v4u*)(hp + (half ? 8 : 0));
        af.q[1] = *(const v4u*)(hp + (half ? 24 : 16));
#pragma unroll
        for (int a = 0; a < 8; a++) {
          const int nt = (a >> 1) * 32 + jt0 + (a & 1);
          ABfrag bf;
          const v4u* bp = (const v4u*)(pBhh + ((size_t)(nt * 16 + kt) * 32 + lane) * 16);
          bf.q[0] = bp[0];
          bf.q[1] = bp[1];
          acc[a] = __builtin_amdgcn_wmma_f32_16x16x32_bf16(
              false, af.v, false, bf.v, (short)0, acc[a], false, false);
        }
      }
      // ---- gates (torch order i,f,g,o), cell state in registers ----------
#pragma unroll
      for (int jt2 = 0; jt2 < 2; jt2++) {
#pragma unroll
        for (int r = 0; r < 8; r++) {
          const float zi = acc[0 + jt2][r], zf = acc[2 + jt2][r];
          const float zg = acc[4 + jt2][r], zo = acc[6 + jt2][r];
          const float cn = sigf(zf) * cst[jt2][r] + sigf(zi) * tanhf(zg);
          cst[jt2][r]  = cn;
          hval[jt2][r] = sigf(zo) * tanhf(cn);
        }
      }
      __syncthreads();   // (1) everyone done reading h(t-1)

      // ---- phase B: publish bf16 h(t), tanh(h(t)); final h,c to global ---
#pragma unroll
      for (int jt2 = 0; jt2 < 2; jt2++) {
        const int j = (jt0 + jt2) * 16 + col;
#pragma unroll
        for (int r = 0; r < 8; r++) {
          const int m = r + half * 8;
          const float hh = hval[jt2][r];
          lds_h[m * 512 + j]  = f2bf(hh);
          lds_th[m * 512 + j] = f2bf(tanhf(hh));
          if (i == 31 && t == 31) {
            out[132096 + (size_t)(b0 + m) * 512 + j] = hh;           // h
            out[148480 + (size_t)(b0 + m) * 512 + j] = cst[jt2][r];  // c
          }
        }
      }
      __syncthreads();   // (2) h(t)/th(t) visible

      // ---- phase C: logits = tanh(h) @ W_lin^T + b_lin (waves 0..8) ------
      if (w < 9) {
        v8f lacc;
#pragma unroll
        for (int r = 0; r < 8; r++) lacc[r] = 0.0f;
#pragma unroll 1
        for (int kt = 0; kt < 16; kt++) {
          ABfrag af;
          const unsigned short* hp = lds_th + col * 512 + kt * 32;
          af.q[0] = *(const v4u*)(hp + (half ? 8 : 0));
          af.q[1] = *(const v4u*)(hp + (half ? 24 : 16));
          ABfrag bf;
          const v4u* bp = (const v4u*)(pBlin + ((size_t)(w * 16 + kt) * 32 + lane) * 16);
          bf.q[0] = bp[0];
          bf.q[1] = bp[1];
          lacc = __builtin_amdgcn_wmma_f32_16x16x32_bf16(
              false, af.v, false, bf.v, (short)0, lacc, false, false);
        }
        const int n = w * 16 + col;
        const float bl = (n < 129) ? b_lin[n] : -1.0e30f;  // pad cols lose argmax
#pragma unroll
        for (int r = 0; r < 8; r++) {
          const int m = r + half * 8;
          const float v = lacc[r] + bl;
          lds_logit[m * 144 + n] = v;
          if (n < 129)
            out[((size_t)(b0 + m) * 32 + t) * 129 + n] = v;  // last iter wins
        }
      }
      __syncthreads();   // (3) logits in LDS

      // ---- phase D: argmax -> next iteration's one-hot indices -----------
      if (tid < 16 && t < 31) {
        float best = lds_logit[tid * 144];
        int bi = 0;
        for (int n = 1; n < 129; n++) {
          const float v = lds_logit[tid * 144 + n];
          if (v > best) { best = v; bi = n; }
        }
        nxt[tid * 32 + t + 1] = bi;
      }
      __syncthreads();   // (4)
    }
  }
}

// ---------------------------------------------------------------------------
extern "C" void kernel_launch(void* const* d_in, const int* in_sizes, int n_in,
                              void* d_out, int out_size, void* d_ws, size_t ws_size,
                              hipStream_t stream) {
  (void)in_sizes; (void)n_in; (void)out_size; (void)ws_size;
  const float* chords = (const float*)d_in[0];
  const float* latent = (const float*)d_in[1];
  const float* W_ih   = (const float*)d_in[2];
  const float* W_hh   = (const float*)d_in[3];
  const float* b_ih   = (const float*)d_in[4];
  const float* b_hh   = (const float*)d_in[5];
  const float* W_lin  = (const float*)d_in[6];
  const float* b_lin  = (const float*)d_in[7];
  float* out = (float*)d_out;

  char* ws = (char*)d_ws;                                     // layout (bytes):
  float*          xw     = (float*)(ws);                      // 0        .. 8,388,608
  unsigned short* pBhh   = (unsigned short*)(ws + 8388608);   // +2,097,152
  unsigned short* pBlin  = (unsigned short*)(ws + 10485760);  // +147,456
  float*          wprevT = (float*)(ws + 10633216);           // +1,056,768

  k_prep_xw <<<8192, 256, 0, stream>>>(chords, latent, W_ih, b_ih, b_hh, xw);
  k_pack_bhh<<<(128 * 16 * 32 + 255) / 256, 256, 0, stream>>>(W_hh, pBhh);
  k_pack_blin<<<(9 * 16 * 32 + 255) / 256, 256, 0, stream>>>(W_lin, pBlin);
  k_wprev   <<<(129 * 2048 + 255) / 256, 256, 0, stream>>>(W_ih, wprevT);
  k_lstm    <<<2, 512, 0, stream>>>(xw, pBhh, pBlin, wprevT, b_lin, out);
}
